// MultiHeadLatentAttention_82059645157463
// MI455X (gfx1250) — matrix operand order repaired and checked by reading, not compiled, not run on hardware
//
#include <hip/hip_runtime.h>

// ---------------------------------------------------------------------------
// MLA forward for MI455X (gfx1250, wave32, WMMA 16x16x32 bf16)
// ---------------------------------------------------------------------------

typedef __attribute__((ext_vector_type(16))) __bf16 v16bf;
typedef __attribute__((ext_vector_type(8)))  __bf16 v8bf;
typedef __attribute__((ext_vector_type(8)))  float  v8f;

#define B_   4
#define S_   1024
#define HID_ 2048
#define NH_  16
#define HD_  128
#define LAT_ 256
#define M_   (B_ * S_)   // 4096

__device__ __forceinline__ __bf16 f2bf(float f) {
    unsigned u = __builtin_bit_cast(unsigned, f);
    u += 0x7FFFu + ((u >> 16) & 1u);              // round-to-nearest-even
    unsigned short h = (unsigned short)(u >> 16);
    return __builtin_bit_cast(__bf16, h);
}
__device__ __forceinline__ float bf2f(__bf16 b) {
    unsigned u = ((unsigned)__builtin_bit_cast(unsigned short, b)) << 16;
    return __builtin_bit_cast(float, u);
}
__device__ __forceinline__ v16bf cat16(v8bf lo, v8bf hi) {
    return __builtin_shufflevector(lo, hi, 0,1,2,3,4,5,6,7,8,9,10,11,12,13,14,15);
}
__device__ __forceinline__ v8f wmma_bf16(v16bf a, v16bf b, v8f c) {
    // D = A(16x32 bf16) * B(32x16 bf16) + C(16x16 f32)
    return __builtin_amdgcn_wmma_f32_16x16x32_bf16(false, a, false, b,
                                                   (short)0, c, false, false);
}

// ---------------------------------------------------------------------------
// Preprocessing: f32 -> bf16 (same layout)
// ---------------------------------------------------------------------------
__global__ void cvt_f32_to_bf16(const float* __restrict__ in,
                                __bf16* __restrict__ out, size_t n) {
    size_t stride = (size_t)gridDim.x * blockDim.x;
    for (size_t t = (size_t)blockIdx.x * blockDim.x + threadIdx.x; t < n; t += stride)
        out[t] = f2bf(in[t]);
}

// f32 [R][C] -> bf16 [C][R], 32x32 LDS-tiled, coalesced both sides.
// R and C must be multiples of 32 (true for all weights here).
__global__ __launch_bounds__(256)
void transpose_f32_bf16(const float* __restrict__ in, __bf16* __restrict__ out,
                        int R, int C) {
    __shared__ float tile[32][33];
    const int tx = threadIdx.x;          // 0..31
    const int ty = threadIdx.y;          // 0..7
    const int c0 = blockIdx.x * 32;      // input column tile
    const int r0 = blockIdx.y * 32;      // input row tile
    #pragma unroll
    for (int j = 0; j < 32; j += 8)
        tile[ty + j][tx] = in[(size_t)(r0 + ty + j) * C + (c0 + tx)];
    __syncthreads();
    #pragma unroll
    for (int j = 0; j < 32; j += 8)
        out[(size_t)(c0 + ty + j) * R + (r0 + tx)] = f2bf(tile[tx][ty + j]);
}

// ---------------------------------------------------------------------------
// GEMM: C[M][N] = A[M][K] (bf16 row-major) * Bt[N][K]^T (bf16, pre-transposed)
// One wave computes a 32x64 tile: 2 A-fragments x 4 B-fragments = 8 WMMAs per
// K-step; every B-fragment feeds two WMMAs. Block = 8 waves = 256 rows.
// MODE 0: bf16 out row-major      MODE 2: f32 out row-major
// MODE 1: bf16 scatter to [B][NH][S][HD]    MODE 3: bf16 scatter to [B][NH][HD][S]
// ---------------------------------------------------------------------------
template <int MODE>
__global__ __launch_bounds__(256)
void gemm_bf16_wmma(const __bf16* __restrict__ A, const __bf16* __restrict__ Bt,
                    void* __restrict__ outp, int M, int N, int K) {
    const int lane = threadIdx.x & 31;
    const int wave = threadIdx.x >> 5;
    const int col  = lane & 15;
    const int half = lane >> 4;
    const int m0 = (blockIdx.y * 8 + wave) * 32;
    const int n0 = blockIdx.x * 64;

    const v8f zf = {0.f,0.f,0.f,0.f,0.f,0.f,0.f,0.f};
    v8f acc[2][4] = { { zf, zf, zf, zf }, { zf, zf, zf, zf } };

    const __bf16* arow0 = A + (size_t)(m0 + col) * K;
    const __bf16* arow1 = A + (size_t)(m0 + 16 + col) * K;
    for (int k0 = 0; k0 < K; k0 += 32) {
        // A fragment: lane<16 -> K = k0+{0..15}; lane>=16 -> +16
        const int kb = k0 + half * 16;
        v16bf av0 = cat16(*(const v8bf*)(arow0 + kb),
                          *(const v8bf*)(arow0 + kb + 8));
        v16bf av1 = cat16(*(const v8bf*)(arow1 + kb),
                          *(const v8bf*)(arow1 + kb + 8));
        __builtin_prefetch(arow0 + kb + 64, 0, 0);   // global_prefetch_b8
        const int kb2 = k0 + half * 16;              // B fragment: K = kb2 + j
        #pragma unroll
        for (int t = 0; t < 4; ++t) {
            const __bf16* brow = Bt + (size_t)(n0 + 16 * t + col) * K + kb2;
            v16bf bv = cat16(*(const v8bf*)brow, *(const v8bf*)(brow + 8));
            acc[0][t] = wmma_bf16(av0, bv, acc[0][t]);
            acc[1][t] = wmma_bf16(av1, bv, acc[1][t]);
        }
    }

    #pragma unroll
    for (int u = 0; u < 2; ++u) {
        #pragma unroll
        for (int t = 0; t < 4; ++t) {
            #pragma unroll
            for (int r = 0; r < 8; ++r) {
                const int m = m0 + 16 * u + r + 8 * half;  // C layout: vgpr r, lane half
                const int n = n0 + 16 * t + col;
                const float v = acc[u][t][r];
                if (MODE == 2) {
                    ((float*)outp)[(size_t)m * N + n] = v;
                } else if (MODE == 0) {
                    ((__bf16*)outp)[(size_t)m * N + n] = f2bf(v);
                } else {
                    const int b = m >> 10, s = m & 1023;    // S_ = 1024
                    const int h = n >> 7,  d = n & 127;     // HD_ = 128
                    const size_t idx = (MODE == 1)
                        ? ((((size_t)b * NH_ + h) * S_ + s) * HD_ + d)
                        : ((((size_t)b * NH_ + h) * HD_ + d) * S_ + s);
                    ((__bf16*)outp)[idx] = f2bf(v);
                }
            }
        }
    }
}

// ---------------------------------------------------------------------------
// RoPE in-place on Q/K stored as bf16 [B][NH][S][HD]
// ---------------------------------------------------------------------------
__global__ void rope_inplace(__bf16* __restrict__ Q, __bf16* __restrict__ Kk) {
    const size_t idx = (size_t)blockIdx.x * blockDim.x + threadIdx.x; // 2^22 total
    const int i  = (int)(idx & 63);
    const int s  = (int)((idx >> 6) & 1023);
    const int h  = (int)((idx >> 16) & 15);
    const int b  = (int)(idx >> 20);
    const size_t base = ((((size_t)b * NH_ + h) * S_) + s) * HD_;
    // inv_freq = theta^(-2i/128) = exp(-i * ln(1e5)/64)
    const float freq = (float)s * __expf(-(float)i * (11.512925464970229f / 64.0f));
    const float c = __cosf(freq), sn = __sinf(freq);
    {
        float t0 = bf2f(Q[base + i]), t1 = bf2f(Q[base + i + 64]);
        Q[base + i]      = f2bf(t0 * c - t1 * sn);
        Q[base + i + 64] = f2bf(t1 * c + t0 * sn);
    }
    {
        float t0 = bf2f(Kk[base + i]), t1 = bf2f(Kk[base + i + 64]);
        Kk[base + i]      = f2bf(t0 * c - t1 * sn);
        Kk[base + i + 64] = f2bf(t1 * c + t0 * sn);
    }
}

// ---------------------------------------------------------------------------
// Flash attention: one wave per (b, h, 16-query tile). D = 128, causal.
// Q/K: [B][NH][S][128] bf16, V: [B][NH][128][S] bf16 (transposed).
// Out: bf16 [B*S][2048], column = h*128 + d.
// ---------------------------------------------------------------------------
__global__ __launch_bounds__(32)
void mla_attn(const __bf16* __restrict__ Qr, const __bf16* __restrict__ Kr,
              const __bf16* __restrict__ Vt, __bf16* __restrict__ Obf) {
    __shared__ __bf16 P[16 * 32];
    const int lane = threadIdx.x & 31;
    const int col = lane & 15, half = lane >> 4;
    const int qt = blockIdx.x & 63;
    const int h  = (blockIdx.x >> 6) & 15;
    const int b  = blockIdx.x >> 10;
    const int q0 = qt * 16;
    const size_t bh = ((size_t)b * NH_ + h) * (size_t)(S_ * HD_);
    const __bf16* Qb = Qr + bh;
    const __bf16* Kb = Kr + bh;
    const __bf16* Vb = Vt + bh;

    // Q A-fragments for the 4 d-chunks of 32
    v16bf qa[4];
    #pragma unroll
    for (int dc = 0; dc < 4; ++dc) {
        const __bf16* p = Qb + (size_t)(q0 + col) * HD_ + dc * 32 + half * 16;
        qa[dc] = cat16(*(const v8bf*)p, *(const v8bf*)(p + 8));
    }

    const v8f zf = {0.f,0.f,0.f,0.f,0.f,0.f,0.f,0.f};
    v8f o[8] = { zf, zf, zf, zf, zf, zf, zf, zf };
    float mrun[8], lrun[8];
    #pragma unroll
    for (int r = 0; r < 8; ++r) { mrun[r] = -1e30f; lrun[r] = 0.f; }

    const float SCALE = 0.08838834764831845f;   // 1/sqrt(128)

    for (int kb0 = 0; kb0 < q0 + 16; kb0 += 32) {
        // ---- scores: two 16x16 tiles over 32 keys ----
        float sc[2][8];
        #pragma unroll
        for (int tn = 0; tn < 2; ++tn) {
            v8f acc = zf;
            const int kn = kb0 + 16 * tn + col;
            const __bf16* krow = Kb + (size_t)kn * HD_ + half * 16;
            #pragma unroll
            for (int dc = 0; dc < 4; ++dc) {
                v16bf kv = cat16(*(const v8bf*)(krow + dc * 32),
                                 *(const v8bf*)(krow + dc * 32 + 8));
                acc = wmma_bf16(qa[dc], kv, acc);
            }
            #pragma unroll
            for (int r = 0; r < 8; ++r) {
                const int qm = q0 + r + 8 * half;
                sc[tn][r] = (kn <= qm) ? acc[r] * SCALE : -1e30f;
            }
        }
        // ---- online softmax (rows live across 16-lane halves) ----
        float mx[8];
        #pragma unroll
        for (int r = 0; r < 8; ++r) mx[r] = fmaxf(sc[0][r], sc[1][r]);
        #pragma unroll
        for (int off = 8; off > 0; off >>= 1)
            #pragma unroll
            for (int r = 0; r < 8; ++r)
                mx[r] = fmaxf(mx[r], __shfl_xor(mx[r], off, 32));
        float corr[8], ps[8];
        #pragma unroll
        for (int r = 0; r < 8; ++r) {
            const float nm = fmaxf(mrun[r], mx[r]);
            corr[r] = __expf(mrun[r] - nm);
            mrun[r] = nm;
            ps[r] = 0.f;
        }
        #pragma unroll
        for (int tn = 0; tn < 2; ++tn)
            #pragma unroll
            for (int r = 0; r < 8; ++r) {
                const float p = __expf(sc[tn][r] - mrun[r]);
                sc[tn][r] = p;
                ps[r] += p;
            }
        #pragma unroll
        for (int off = 8; off > 0; off >>= 1)
            #pragma unroll
            for (int r = 0; r < 8; ++r)
                ps[r] += __shfl_xor(ps[r], off, 32);
        #pragma unroll
        for (int r = 0; r < 8; ++r) lrun[r] = lrun[r] * corr[r] + ps[r];
        #pragma unroll
        for (int dn = 0; dn < 8; ++dn)
            #pragma unroll
            for (int r = 0; r < 8; ++r) o[dn][r] *= corr[r];

        // ---- reshape P (C-layout f32) -> A-fragment via LDS ----
        #pragma unroll
        for (int tn = 0; tn < 2; ++tn)
            #pragma unroll
            for (int r = 0; r < 8; ++r)
                P[(r + 8 * half) * 32 + 16 * tn + col] = f2bf(sc[tn][r]);
        asm volatile("s_wait_dscnt 0" ::: "memory");   // single wave: DS in-order
        const __bf16* pp = P + col * 32 + half * 16;
        v16bf pa = cat16(*(const v8bf*)pp, *(const v8bf*)(pp + 8));
        asm volatile("" ::: "memory");

        // ---- O += P * V  (8 d-tiles of 16) ----
        #pragma unroll
        for (int dn = 0; dn < 8; ++dn) {
            const __bf16* vrow = Vb + (size_t)(16 * dn + col) * S_ + kb0 + half * 16;
            v16bf vv = cat16(*(const v8bf*)vrow, *(const v8bf*)(vrow + 8));
            o[dn] = wmma_bf16(pa, vv, o[dn]);
        }
    }

    // ---- finalize ----
    #pragma unroll
    for (int dn = 0; dn < 8; ++dn)
        #pragma unroll
        for (int r = 0; r < 8; ++r) {
            const int m = r + 8 * half;
            const size_t row = (size_t)b * S_ + q0 + m;
            Obf[row * HID_ + h * HD_ + 16 * dn + col] = f2bf(o[dn][r] / lrun[r]);
        }
}

// ---------------------------------------------------------------------------
// Host-side launcher
// ---------------------------------------------------------------------------
extern "C" void kernel_launch(void* const* d_in, const int* in_sizes, int n_in,
                              void* d_out, int out_size, void* d_ws, size_t ws_size,
                              hipStream_t stream) {
    const float* x    = (const float*)d_in[0];
    const float* wqd  = (const float*)d_in[1];   // [2048][256]
    const float* wkvd = (const float*)d_in[2];   // [2048][256]
    const float* wqu  = (const float*)d_in[3];   // [256][2048]
    const float* wku  = (const float*)d_in[4];   // [256][2048]
    const float* wvu  = (const float*)d_in[5];   // [256][2048]
    const float* wo   = (const float*)d_in[6];   // [2048][2048]
    float* out = (float*)d_out;

    char* ws = (char*)d_ws;
    size_t off = 0;
    auto alloc = [&](size_t bytes) { char* p = ws + off; off += bytes; return p; };

    __bf16* xbf   = (__bf16*)alloc((size_t)M_ * HID_ * 2);     // 16 MB
    __bf16* wqdT  = (__bf16*)alloc((size_t)LAT_ * HID_ * 2);   // [256][2048]
    __bf16* wkvdT = (__bf16*)alloc((size_t)LAT_ * HID_ * 2);
    __bf16* wquT  = (__bf16*)alloc((size_t)HID_ * LAT_ * 2);   // [2048][256]
    __bf16* wkuT  = (__bf16*)alloc((size_t)HID_ * LAT_ * 2);
    __bf16* wvuT  = (__bf16*)alloc((size_t)HID_ * LAT_ * 2);
    __bf16* woT   = (__bf16*)alloc((size_t)HID_ * HID_ * 2);   // [2048][2048]
    __bf16* qlat  = (__bf16*)alloc((size_t)M_ * LAT_ * 2);
    __bf16* kvlat = (__bf16*)alloc((size_t)M_ * LAT_ * 2);
    __bf16* Qr    = (__bf16*)alloc((size_t)M_ * HID_ * 2);     // [B][NH][S][HD]
    __bf16* Kr    = (__bf16*)alloc((size_t)M_ * HID_ * 2);
    __bf16* Vt    = (__bf16*)alloc((size_t)M_ * HID_ * 2);     // [B][NH][HD][S]
    __bf16* Obf   = (__bf16*)alloc((size_t)M_ * HID_ * 2);
    (void)ws_size; (void)in_sizes; (void)n_in; (void)out_size;

    // 1) precision conversion / transposition
    cvt_f32_to_bf16<<<2048, 256, 0, stream>>>(x, xbf, (size_t)M_ * HID_);
    {
        dim3 blk(32, 8);
        transpose_f32_bf16<<<dim3(LAT_ / 32, HID_ / 32), blk, 0, stream>>>(wqd,  wqdT,  HID_, LAT_);
        transpose_f32_bf16<<<dim3(LAT_ / 32, HID_ / 32), blk, 0, stream>>>(wkvd, wkvdT, HID_, LAT_);
        transpose_f32_bf16<<<dim3(HID_ / 32, LAT_ / 32), blk, 0, stream>>>(wqu,  wquT,  LAT_, HID_);
        transpose_f32_bf16<<<dim3(HID_ / 32, LAT_ / 32), blk, 0, stream>>>(wku,  wkuT,  LAT_, HID_);
        transpose_f32_bf16<<<dim3(HID_ / 32, LAT_ / 32), blk, 0, stream>>>(wvu,  wvuT,  LAT_, HID_);
        transpose_f32_bf16<<<dim3(HID_ / 32, HID_ / 32), blk, 0, stream>>>(wo,   woT,   HID_, HID_);
    }

    // 2) down projections (M=4096, N=256, K=2048); 256 rows x 64 cols per block
    gemm_bf16_wmma<0><<<dim3(LAT_ / 64, M_ / 256), 256, 0, stream>>>(
        xbf, wqdT, qlat, M_, LAT_, HID_);
    gemm_bf16_wmma<0><<<dim3(LAT_ / 64, M_ / 256), 256, 0, stream>>>(
        xbf, wkvdT, kvlat, M_, LAT_, HID_);

    // 3) up projections (M=4096, N=2048, K=256) with layout scatter
    gemm_bf16_wmma<1><<<dim3(HID_ / 64, M_ / 256), 256, 0, stream>>>(
        qlat, wquT, Qr, M_, HID_, LAT_);
    gemm_bf16_wmma<1><<<dim3(HID_ / 64, M_ / 256), 256, 0, stream>>>(
        kvlat, wkuT, Kr, M_, HID_, LAT_);
    gemm_bf16_wmma<3><<<dim3(HID_ / 64, M_ / 256), 256, 0, stream>>>(
        kvlat, wvuT, Vt, M_, HID_, LAT_);

    // 4) RoPE on Q and K (B*NH*S*64 = 2^22 pairs)
    rope_inplace<<<(B_ * NH_ * S_ * 64) / 256, 256, 0, stream>>>(Qr, Kr);

    // 5) causal flash attention: one wave per (b, h, 16-query tile)
    mla_attn<<<B_ * NH_ * (S_ / 16), 32, 0, stream>>>(Qr, Kr, Vt, Obf);

    // 6) output projection (M=4096, N=2048, K=2048) -> fp32 out
    gemm_bf16_wmma<2><<<dim3(HID_ / 64, M_ / 256), 256, 0, stream>>>(
        Obf, woT, out, M_, HID_, HID_);
}
